// DynamicChunkingDownsampler_42829413875991
// MI455X (gfx1250) — matrix operand order repaired and checked by reading, not compiled, not run on hardware
//
#include <hip/hip_runtime.h>

// ---------------------------------------------------------------------------
// Problem constants
// ---------------------------------------------------------------------------
#define BB   4
#define LL   4096
#define DD   2048
#define DQK_ 2048
#define GM   (BB * LL)        // 16384
#define GN   (2 * DQK_)       // 4096
#define GK   DD               // 2048

typedef __bf16 bf16;
typedef bf16  v16bf __attribute__((ext_vector_type(16)));
typedef bf16  v8bf  __attribute__((ext_vector_type(8)));
typedef bf16  v4bf  __attribute__((ext_vector_type(4)));
typedef float v8f   __attribute__((ext_vector_type(8)));

// ---------------------------------------------------------------------------
// Kernel 0a: f32 -> bf16 conversion (vectorized float4 -> 4x bf16)
// ---------------------------------------------------------------------------
__global__ __launch_bounds__(256) void cvt_bf16_kernel(
    const float* __restrict__ in, unsigned short* __restrict__ outp, int n4)
{
    int i = blockIdx.x * 256 + threadIdx.x;
    if (i >= n4) return;
    float4 v = reinterpret_cast<const float4*>(in)[i];
    v4bf o;
    o[0] = (bf16)v.x; o[1] = (bf16)v.y; o[2] = (bf16)v.z; o[3] = (bf16)v.w;
    reinterpret_cast<v4bf*>(outp)[i] = o;
}

// ---------------------------------------------------------------------------
// Kernel 0b: f32 -> bf16 conversion + transpose for W.
// W[k][n] (f32, row-major) -> Wt[n][k] (bf16, column-major of W).
// 32x32 LDS-tiled so both global sides stay coalesced.
// ---------------------------------------------------------------------------
__global__ __launch_bounds__(256) void cvt_transpose_bf16_kernel(
    const float* __restrict__ W, unsigned short* __restrict__ Wt)
{
    __shared__ bf16 tile[32][33];            // [n][k], padded
    bf16* t = &tile[0][0];
    int k0 = blockIdx.y * 32;
    int n0 = blockIdx.x * 32;
    int tx = threadIdx.x & 7;                // 8 groups of 4 along fast dim
    int ty = threadIdx.x >> 3;               // 32 rows
    // read W[k0+ty][n0 + tx*4 .. +4)
    float4 v = *reinterpret_cast<const float4*>(&W[(size_t)(k0 + ty) * GN + n0 + tx * 4]);
    t[(tx * 4 + 0) * 33 + ty] = (bf16)v.x;
    t[(tx * 4 + 1) * 33 + ty] = (bf16)v.y;
    t[(tx * 4 + 2) * 33 + ty] = (bf16)v.z;
    t[(tx * 4 + 3) * 33 + ty] = (bf16)v.w;
    __syncthreads();
    // write Wt[n0+ty][k0 + tx*4 .. +4)
    v4bf o;
    o[0] = t[ty * 33 + tx * 4 + 0];
    o[1] = t[ty * 33 + tx * 4 + 1];
    o[2] = t[ty * 33 + tx * 4 + 2];
    o[3] = t[ty * 33 + tx * 4 + 3];
    *reinterpret_cast<v4bf*>(&Wt[(size_t)(n0 + ty) * GK + k0 + tx * 4]) = o;
}

// ---------------------------------------------------------------------------
// Kernel 1: bf16 WMMA GEMM  C[GM,GN] = A[GM,GK] * W[GK,GN]  (f32 accumulate)
// Block tile 128x128, BK=32; 256 threads = 8 waves, each wave 32x64 (2x4 WMMA).
// Both tiles are K-contiguous in global (A row-major, Wt = W transposed), so
// both stream through global_load_async_to_lds_b128 into padded, double-
// buffered LDS (ASYNCcnt-tracked), with one barrier per K-step.
// ---------------------------------------------------------------------------
__global__ __launch_bounds__(256) void gemm_bf16_wmma_kernel(
    const unsigned short* __restrict__ Abf,
    const unsigned short* __restrict__ Wtbf,
    float* __restrict__ C)
{
    constexpr int BM = 128, BN = 128, BK = 32, LDT = BK + 8; // 40 bf16 row stride
    __shared__ bf16 As[2][BM * LDT];   // [buf][row][k]
    __shared__ bf16 Bs[2][BN * LDT];   // [buf][n][k]

    const bf16* A  = reinterpret_cast<const bf16*>(Abf);
    const bf16* Wt = reinterpret_cast<const bf16*>(Wtbf);

    const int tid  = threadIdx.x;
    const int bm   = blockIdx.y * BM;
    const int bn   = blockIdx.x * BN;
    const int wid  = tid >> 5;
    const int lane = tid & 31;
    const int wm   = (wid & 3) * 32;   // wave row offset in block tile
    const int wn   = (wid >> 2) * 64;  // wave col offset in block tile
    const int l15  = lane & 15;
    const int lh   = lane >> 4;        // lane half (0/1)

    // Each tile = 512 chunks of 16B; each thread issues 2 async copies per tile.
    auto asyncLoadTile = [&](const bf16* gbase, bf16* lbase, int kt) {
#pragma unroll
        for (int i = 0; i < 2; i++) {
            int ch = tid * 2 + i;
            int r  = ch >> 2;              // tile row (4 chunks per 32-elem row)
            int ko = (ch & 3) * 8;         // k offset (bf16 elems)
            const bf16* g = &gbase[(size_t)r * GK + kt + ko];
            unsigned lds  = (unsigned)(size_t)(void*)&lbase[r * LDT + ko];
            asm volatile("global_load_async_to_lds_b128 %0, %1, off"
                         :: "v"(lds),
                            "v"((unsigned long long)(size_t)g)
                         : "memory");
        }
    };
    auto loadTiles = [&](int kt, int buf) {
        asyncLoadTile(A  + (size_t)bm * GK, &As[buf][0], kt);
        asyncLoadTile(Wt + (size_t)bn * GK, &Bs[buf][0], kt);
    };

    v8f zero = {0.f, 0.f, 0.f, 0.f, 0.f, 0.f, 0.f, 0.f};
    v8f acc[2][4];
#pragma unroll
    for (int mi = 0; mi < 2; mi++)
#pragma unroll
        for (int ni = 0; ni < 4; ni++) acc[mi][ni] = zero;

    // Prologue: fill buffer 0
    loadTiles(0, 0);
    asm volatile("s_wait_asynccnt 0x0" ::: "memory");
    __syncthreads();

    int buf = 0;
    for (int kt = 0; kt < GK; kt += BK) {
        const int  nxt     = buf ^ 1;
        const bool hasNext = (kt + BK < GK);
        if (hasNext)                      // overlap next fill with this compute
            loadTiles(kt + BK, nxt);

        // A fragment (16x32 bf16): lane<16 K {0..7,16..23}; lane>=16 {8..15,24..31}
        v16bf afrag[2];
#pragma unroll
        for (int mi = 0; mi < 2; mi++) {
            int row = wm + mi * 16 + l15;
            int kb  = lh * 8;
            v8bf lo = *reinterpret_cast<const v8bf*>(&As[buf][row * LDT + kb]);
            v8bf hi = *reinterpret_cast<const v8bf*>(&As[buf][row * LDT + kb + 16]);
            afrag[mi] = __builtin_shufflevector(lo, hi,
                0,1,2,3,4,5,6,7,8,9,10,11,12,13,14,15);
        }
        // B fragment (32x16 bf16): lane<16 K {0..7,16..23}; lane>=16 {8..15,24..31}
        v16bf bfrag[4];
#pragma unroll
        for (int ni = 0; ni < 4; ni++) {
            int col = wn + ni * 16 + l15;
            int kb  = lh * 8;
            v8bf lo = *reinterpret_cast<const v8bf*>(&Bs[buf][col * LDT + kb]);
            v8bf hi = *reinterpret_cast<const v8bf*>(&Bs[buf][col * LDT + kb + 16]);
            bfrag[ni] = __builtin_shufflevector(lo, hi,
                0,1,2,3,4,5,6,7,8,9,10,11,12,13,14,15);
        }
#pragma unroll
        for (int mi = 0; mi < 2; mi++)
#pragma unroll
            for (int ni = 0; ni < 4; ni++)
                acc[mi][ni] = __builtin_amdgcn_wmma_f32_16x16x32_bf16(
                    false, afrag[mi], false, bfrag[ni],
                    (short)0, acc[mi][ni], false, false);

        if (hasNext) {
            asm volatile("s_wait_asynccnt 0x0" ::: "memory");  // own copies done
            __syncthreads();   // collective: nxt fully written, cur fully read
        }
        buf = nxt;
    }

    // C/D layout: VGPR r, lanes 0-15 -> (M=r, N=lane); lanes 16-31 -> (M=r+8, N=lane-16)
#pragma unroll
    for (int mi = 0; mi < 2; mi++) {
        int row0 = bm + wm + mi * 16 + lh * 8;
#pragma unroll
        for (int ni = 0; ni < 4; ni++) {
            int col = bn + wn + ni * 16 + l15;
#pragma unroll
            for (int r = 0; r < 8; r++)
                C[(size_t)(row0 + r) * GN + col] = acc[mi][ni][r];
        }
    }
}

// ---------------------------------------------------------------------------
// Kernel 2: cosine similarity -> probs, boundary.  One block per (b,t).
// ---------------------------------------------------------------------------
__global__ __launch_bounds__(256) void cos_probs_kernel(
    const float* __restrict__ qk, const float* __restrict__ start_key,
    float* __restrict__ probs, int* __restrict__ bnd)
{
    __shared__ float sd[256], sq[256], sk[256];
    int idx = blockIdx.x;               // b*L + t
    int t   = idx & (LL - 1);
    int tid = threadIdx.x;
    const float* q  = qk + (size_t)idx * GN;
    const float* kp = (t == 0) ? start_key
                               : (qk + (size_t)(idx - 1) * GN + DQK_);
    float dot = 0.f, qq = 0.f, kk = 0.f;
    for (int i = tid; i < DQK_; i += 256) {
        float qv = q[i], kv = kp[i];
        dot = fmaf(qv, kv, dot);
        qq  = fmaf(qv, qv, qq);
        kk  = fmaf(kv, kv, kk);
    }
    sd[tid] = dot; sq[tid] = qq; sk[tid] = kk;
    __syncthreads();
    for (int s = 128; s > 0; s >>= 1) {
        if (tid < s) { sd[tid] += sd[tid + s]; sq[tid] += sq[tid + s]; sk[tid] += sk[tid + s]; }
        __syncthreads();
    }
    if (tid == 0) {
        float cs = sd[0] / (fmaxf(sqrtf(sq[0]), 1e-8f) * fmaxf(sqrtf(sk[0]), 1e-8f));
        float p  = (1.0f - cs) * 0.5f;
        probs[idx] = p;
        bnd[idx]   = (p > 0.5f || t == 0) ? 1 : 0;
    }
}

// ---------------------------------------------------------------------------
// Kernel 3: per-batch jagged pack (stable compaction) + chunk_id. 1 block / b.
// ---------------------------------------------------------------------------
__global__ __launch_bounds__(1024) void pack_kernel(
    const int* __restrict__ bnd, const float* __restrict__ probs,
    int* __restrict__ pos, float* __restrict__ bp, float* __restrict__ gate,
    int* __restrict__ cid, int* __restrict__ nck)
{
    int b   = blockIdx.x;
    int tid = threadIdx.x;
    __shared__ int ssum[1024];
    int base = b * LL;
    int v[4]; int s = 0;
#pragma unroll
    for (int j = 0; j < 4; j++) { v[j] = bnd[base + tid * 4 + j]; s += v[j]; }
    ssum[tid] = s;
    __syncthreads();
    for (int off = 1; off < 1024; off <<= 1) {
        int add = (tid >= off) ? ssum[tid - off] : 0;
        __syncthreads();
        ssum[tid] += add;
        __syncthreads();
    }
    int c = ssum[tid] - s;   // exclusive prefix before this thread's segment
#pragma unroll
    for (int j = 0; j < 4; j++) {
        int t = tid * 4 + j;
        if (v[j]) {
            float p = probs[base + t];
            pos[base + c]  = t;
            bp[base + c]   = p;
            gate[base + c] = 1.0f - p;
            c++;
        }
        cid[base + t] = c - 1;   // inclusive cumsum - 1 (>=0 since t=0 boundary)
    }
    if (tid == 1023) nck[b] = ssum[1023];
}

// ---------------------------------------------------------------------------
// Kernel 4: gated linear recurrence over packed chunks. Parallel over (b,d).
// Padding rows carry forward last h (gate=1, x=0 in reference).
// ---------------------------------------------------------------------------
__global__ __launch_bounds__(256) void gscan_kernel(
    const float* __restrict__ tokens,
    const int* __restrict__ pos, const float* __restrict__ bp,
    const float* __restrict__ gate, const int* __restrict__ nck,
    float* __restrict__ out_s)
{
    __shared__ int   sPos[LL];
    __shared__ float sG[LL];
    __shared__ float sB[LL];
    int b = blockIdx.y;
    int d = blockIdx.x * 256 + threadIdx.x;
    int n = nck[b];
    for (int i = threadIdx.x; i < n; i += 256) {
        sPos[i] = pos[b * LL + i];
        sG[i]   = gate[b * LL + i];
        sB[i]   = bp[b * LL + i];
    }
    __syncthreads();
    float h = 0.f;
    size_t rb = ((size_t)b * LL) * DD + d;
    int c = 0;
    for (; c < n; ++c) {
        float tok = tokens[rb + (size_t)sPos[c] * DD];
        h = fmaf(sG[c], h, sB[c] * tok);
        out_s[rb + (size_t)c * DD] = h;
    }
    for (; c < LL; ++c)
        out_s[rb + (size_t)c * DD] = h;   // carry-forward padding
}

// ---------------------------------------------------------------------------
// Kernel 5: upsample gather + aux scalar. One block per (b,t), float4 rows.
// ---------------------------------------------------------------------------
__global__ __launch_bounds__(512) void upsample_kernel(
    const float* __restrict__ out_s, const int* __restrict__ cid,
    float* __restrict__ out_u, float* __restrict__ aux)
{
    int idx = blockIdx.x;               // b*L + t
    int b   = idx >> 12;                // L = 4096
    int c   = cid[idx];
    const float4* src = reinterpret_cast<const float4*>(
        out_s + ((size_t)(b * LL) + c) * DD);
    float4* dst = reinterpret_cast<float4*>(out_u + (size_t)idx * DD);
    dst[threadIdx.x] = src[threadIdx.x];
    if (idx == 0 && threadIdx.x == 0) aux[0] = 0.f;
}

// ---------------------------------------------------------------------------
// Host entry
// ---------------------------------------------------------------------------
extern "C" void kernel_launch(void* const* d_in, const int* in_sizes, int n_in,
                              void* d_out, int out_size, void* d_ws, size_t ws_size,
                              hipStream_t stream)
{
    const float* tokens    = (const float*)d_in[0];
    const float* W_qk      = (const float*)d_in[1];
    const float* start_key = (const float*)d_in[2];
    float* out = (float*)d_out;

    char*  ws  = (char*)d_ws;
    size_t off = 0;
    float*          qk    = (float*)(ws + off);          off += (size_t)GM * GN * 4;
    unsigned short* Abf   = (unsigned short*)(ws + off); off += (size_t)GM * GK * 2;
    unsigned short* Wtbf  = (unsigned short*)(ws + off); off += (size_t)GK * GN * 2;
    float*          probs = (float*)(ws + off);          off += (size_t)GM * 4;
    int*            bnd   = (int*)(ws + off);            off += (size_t)GM * 4;
    int*            pos   = (int*)(ws + off);            off += (size_t)GM * 4;
    float*          bp    = (float*)(ws + off);          off += (size_t)GM * 4;
    float*          gate  = (float*)(ws + off);          off += (size_t)GM * 4;
    int*            cid   = (int*)(ws + off);            off += (size_t)GM * 4;
    int*            nck   = (int*)(ws + off);            off += 64;

    // 0) tokens f32 -> bf16; W f32 -> bf16 transposed (Wt[n][k])
    {
        int n4a = GM * GK / 4;
        cvt_bf16_kernel<<<(n4a + 255) / 256, 256, 0, stream>>>(tokens, Abf, n4a);
        cvt_transpose_bf16_kernel<<<dim3(GN / 32, GK / 32), 256, 0, stream>>>(W_qk, Wtbf);
    }
    // 1) qk = tokens @ W_qk  via v_wmma_f32_16x16x32_bf16 + async global->LDS
    gemm_bf16_wmma_kernel<<<dim3(GN / 128, GM / 128), 256, 0, stream>>>(Abf, Wtbf, qk);
    // 2) cosine similarity -> probs, boundary
    cos_probs_kernel<<<GM, 256, 0, stream>>>(qk, start_key, probs, bnd);
    // 3) jagged pack per batch
    pack_kernel<<<BB, 1024, 0, stream>>>(bnd, probs, pos, bp, gate, cid, nck);
    // 4) gated scan -> smoothed (first output region)
    gscan_kernel<<<dim3(DD / 256, BB), 256, 0, stream>>>(tokens, pos, bp, gate, nck, out);
    // 5) upsample -> second output region; aux scalar at the end
    upsample_kernel<<<GM, 512, 0, stream>>>(
        out, cid, out + (size_t)GM * DD, out + 2ull * GM * DD);
}
